// FixedEmbedderNN_39367670235826
// MI455X (gfx1250) — compile-verified, matrix operand order checked by reading, not run on hardware
//
#include <hip/hip_runtime.h>

typedef __attribute__((ext_vector_type(16))) _Float16 v16h;
typedef __attribute__((ext_vector_type(8)))  float    v8f;

#define N_CAT 20
#define N_NUM 20
#define NFEAT 40
#define EMB_D 32
#define HID   128
#define POST  1280

// workspace layout (in halves)
#define EMB_OFF   0
#define EMB_SZ    (N_CAT*50*EMB_D)        // 32000
#define WIN_OFF   (EMB_OFF+EMB_SZ)        // 32000
#define WIN_SZ    (HID*POST)              // 163840  (W_in^T : [n][k])
#define W1_OFF    (WIN_OFF+WIN_SZ)        // 195840
#define W1_SZ     (2*256*HID)             // 65536   (W1^T : [l][n][k])
#define W2_OFF    (W1_OFF+W1_SZ)          // 261376
#define W2_SZ     (2*HID*256)             // 65536   (W2^T : [l][n][k])
#define WOUT_OFF  (W2_OFF+W2_SZ)          // 326912
#define WOUT_SZ   (HID*HID)               // 16384   (W_out^T : [n][k])
#define WS_TOTAL  (WOUT_OFF+WOUT_SZ)      // 343296 halves = 686,592 B

#define WMMA(a,b,c) __builtin_amdgcn_wmma_f32_16x16x32_f16(false,(a),false,(b),(short)0,(c),false,false)

union FragU {
  v16h     v;
  uint4    u[2];
  _Float16 h[16];
};

__device__ __forceinline__ v16h ldfrag(const _Float16* p0, const _Float16* p1) {
  FragU f;
  f.u[0] = *(const uint4*)p0;
  f.u[1] = *(const uint4*)p1;
  return f.v;
}

__device__ __forceinline__ v8f vzero() {
  v8f z = {0.f,0.f,0.f,0.f,0.f,0.f,0.f,0.f};
  return z;
}

// ---------------- prep: fp32 -> f16, pre-transposed weights into d_ws ----------------
__global__ void prep_kernel(const float* __restrict__ emb,
                            const float* __restrict__ win,
                            const float* __restrict__ w1,
                            const float* __restrict__ w2,
                            const float* __restrict__ wout,
                            _Float16* __restrict__ ws) {
  int i = blockIdx.x * blockDim.x + threadIdx.x;
  if (i >= WS_TOTAL) return;
  float v;
  if (i < WIN_OFF) {                       // embedding tables, plain convert
    v = emb[i];
  } else if (i < W1_OFF) {                 // W_in^T [n][k], src [k][n]
    int j = i - WIN_OFF; int n = j / POST; int k = j % POST;
    v = win[k * HID + n];
  } else if (i < W2_OFF) {                 // W1^T [l][n][k], src [l][k][n], k=128 n=256
    int j = i - W1_OFF; int l = j >> 15; int r = j & 32767;
    int n = r >> 7; int k = r & 127;
    v = w1[l * 32768 + k * 256 + n];
  } else if (i < WOUT_OFF) {               // W2^T [l][n][k], src [l][k][n], k=256 n=128
    int j = i - W2_OFF; int l = j >> 15; int r = j & 32767;
    int n = r >> 8; int k = r & 255;
    v = w2[l * 32768 + k * 128 + n];
  } else {                                 // W_out^T [n][k], src [k][n]
    int j = i - WOUT_OFF; int n = j >> 7; int k = j & 127;
    v = wout[k * HID + n];
  }
  ws[i] = (_Float16)v;
}

// ---------------- fused main kernel ----------------
__global__ __launch_bounds__(256, 1)
void embnn_main(const float* __restrict__ x,
                const float* __restrict__ Wnum, const float* __restrict__ Bnum,
                const float* __restrict__ b_in,
                const float* __restrict__ b1,  const float* __restrict__ b2,
                const float* __restrict__ ln_g, const float* __restrict__ ln_b,
                const float* __restrict__ b_out,
                const _Float16* __restrict__ ws,
                float* __restrict__ out) {
  __shared__ _Float16 hbuf[8 * 16 * HID];   // 32 KB : h activations (f16, row-major)
  __shared__ _Float16 gbuf[8 * 16 * 256];   // 64 KB : g activations (f16, row-major)

  const int tid  = threadIdx.x;
  const int lane = tid & 31;
  const int wave = tid >> 5;
  const int hs   = lane >> 4;     // half-wave select
  const int lr   = lane & 15;
  const int rowBase = blockIdx.x * 128 + wave * 16;
  const int mrow = rowBase + lr;  // row this lane loads for A fragments

  const _Float16* embT = ws + EMB_OFF;
  const _Float16* winT = ws + WIN_OFF;
  const _Float16* w1T  = ws + W1_OFF;
  const _Float16* w2T  = ws + W2_OFF;
  const _Float16* woT  = ws + WOUT_OFF;

  _Float16* hme = hbuf + wave * (16 * HID);
  _Float16* gme = gbuf + wave * (16 * 256);

  // per-lane K-group offsets inside a 32-wide K chunk (matches ISA A/B 16-bit layout)
  const int g0 = hs * 8;
  const int g1 = 16 + hs * 8;

  // ===== stage 1: fused embedding/numeric-linear + (N,1280)@W_in, one K=32 step per feature =====
  v8f acc[8];
  #pragma unroll
  for (int nt = 0; nt < 8; ++nt) acc[nt] = vzero();

  #pragma unroll 1
  for (int f = 0; f < N_CAT; ++f) {
    float xv = x[mrow * NFEAT + f];
    int idx = (int)xv;
    const _Float16* ep = embT + (f * 50 + idx) * EMB_D;
    v16h a = ldfrag(ep + g0, ep + g1);
    #pragma unroll
    for (int nt = 0; nt < 8; ++nt) {
      const _Float16* bp = winT + (nt * 16 + lr) * POST + f * EMB_D;
      v16h b = ldfrag(bp + g0, bp + g1);
      acc[nt] = WMMA(a, b, acc[nt]);
    }
  }
  #pragma unroll 1
  for (int f = N_CAT; f < NFEAT; ++f) {
    float xv = x[mrow * NFEAT + f];
    const float* wn = Wnum + (f - N_CAT) * EMB_D;
    const float* bn = Bnum + (f - N_CAT) * EMB_D;
    FragU a;
    #pragma unroll
    for (int t = 0; t < 8; ++t) {
      a.h[t]     = (_Float16)(xv * wn[g0 + t] + bn[g0 + t]);
      a.h[8 + t] = (_Float16)(xv * wn[g1 + t] + bn[g1 + t]);
    }
    #pragma unroll
    for (int nt = 0; nt < 8; ++nt) {
      const _Float16* bp = winT + (nt * 16 + lr) * POST + f * EMB_D;
      v16h b = ldfrag(bp + g0, bp + g1);
      acc[nt] = WMMA(a.v, b, acc[nt]);
    }
  }
  // + b_in, stash h as f16 in LDS (row-major [16][128] per wave)
  #pragma unroll
  for (int nt = 0; nt < 8; ++nt) {
    float bv = b_in[nt * 16 + lr];
    #pragma unroll
    for (int r = 0; r < 8; ++r)
      hme[(r + hs * 8) * HID + nt * 16 + lr] = (_Float16)(acc[nt][r] + bv);
  }
  __syncthreads();

  // ===== 2 hidden layers =====
  #pragma unroll 1
  for (int l = 0; l < 2; ++l) {
    const _Float16* w1l = w1T + l * 256 * HID;
    const _Float16* w2l = w2T + l * HID * 256;

    // preload the 4 A fragments of h (K=128), reused by both N-tile batches
    v16h ah[4];
    #pragma unroll
    for (int kc = 0; kc < 4; ++kc) {
      const _Float16* ap = hme + lr * HID + kc * 32;
      ah[kc] = ldfrag(ap + g0, ap + g1);
    }

    // GEMM1: g = h @ W1 + b1   (256 cols, two batches of 8 N-tiles)
    #pragma unroll 1
    for (int gh = 0; gh < 2; ++gh) {
      v8f acg[8];
      #pragma unroll
      for (int nt = 0; nt < 8; ++nt) acg[nt] = vzero();
      #pragma unroll
      for (int kc = 0; kc < 4; ++kc) {
        #pragma unroll
        for (int nt = 0; nt < 8; ++nt) {
          int n = (gh * 8 + nt) * 16 + lr;
          const _Float16* bp = w1l + n * HID + kc * 32;
          v16h b = ldfrag(bp + g0, bp + g1);
          acg[nt] = WMMA(ah[kc], b, acg[nt]);
        }
      }
      #pragma unroll
      for (int nt = 0; nt < 8; ++nt) {
        int n = (gh * 8 + nt) * 16 + lr;
        float bv = b1[l * 256 + n];
        #pragma unroll
        for (int r = 0; r < 8; ++r)
          gme[(r + hs * 8) * 256 + n] = (_Float16)(acg[nt][r] + bv);
      }
    }
    __syncthreads();

    // GEMM2: h2 = g @ W2 + b2  (K=256)
    v8f ach[8];
    #pragma unroll
    for (int nt = 0; nt < 8; ++nt) ach[nt] = vzero();
    #pragma unroll
    for (int kc = 0; kc < 8; ++kc) {
      const _Float16* ap = gme + lr * 256 + kc * 32;
      v16h a = ldfrag(ap + g0, ap + g1);
      #pragma unroll
      for (int nt = 0; nt < 8; ++nt) {
        const _Float16* bp = w2l + (nt * 16 + lr) * 256 + kc * 32;
        v16h b = ldfrag(bp + g0, bp + g1);
        ach[nt] = WMMA(a, b, ach[nt]);
      }
    }

    // + b2 (same n for all 8 accumulator elements of a tile)
    #pragma unroll
    for (int nt = 0; nt < 8; ++nt) {
      float bv = b2[l * HID + nt * 16 + lr];
      #pragma unroll
      for (int r = 0; r < 8; ++r) ach[nt][r] += bv;
    }

    // ===== layernorm entirely in registers =====
    // row m = r + hs*8 is striped across the 16 lanes of one half-wave;
    // sum over n = 8 elementwise v8f adds + 4 shfl_xor steps (stay within half-wave)
    v8f ps = vzero(), qs = vzero();
    #pragma unroll
    for (int nt = 0; nt < 8; ++nt) {
      ps += ach[nt];
      qs += ach[nt] * ach[nt];
    }
    #pragma unroll
    for (int msk = 1; msk < 16; msk <<= 1) {
      #pragma unroll
      for (int r = 0; r < 8; ++r) {
        ps[r] += __shfl_xor(ps[r], msk, 32);
        qs[r] += __shfl_xor(qs[r], msk, 32);
      }
    }
    v8f mu, rstd;
    #pragma unroll
    for (int r = 0; r < 8; ++r) {
      float m  = ps[r] * (1.0f / HID);
      float vv = qs[r] * (1.0f / HID) - m * m;
      mu[r]   = m;
      rstd[r] = rsqrtf(vv + 1e-5f);
    }
    // normalize + affine, write f16 straight into hbuf for the next GEMM
    #pragma unroll
    for (int nt = 0; nt < 8; ++nt) {
      int n = nt * 16 + lr;
      float gam = ln_g[l * HID + n];
      float bet = ln_b[l * HID + n];
      #pragma unroll
      for (int r = 0; r < 8; ++r) {
        float y = (ach[nt][r] - mu[r]) * rstd[r] * gam + bet;
        hme[(r + hs * 8) * HID + n] = (_Float16)y;
      }
    }
    __syncthreads();
  }

  // ===== output GEMM: out = h @ W_out + b_out =====
  v8f aco[8];
  #pragma unroll
  for (int nt = 0; nt < 8; ++nt) aco[nt] = vzero();
  #pragma unroll
  for (int kc = 0; kc < 4; ++kc) {
    const _Float16* ap = hme + lr * HID + kc * 32;
    v16h a = ldfrag(ap + g0, ap + g1);
    #pragma unroll
    for (int nt = 0; nt < 8; ++nt) {
      const _Float16* bp = woT + (nt * 16 + lr) * HID + kc * 32;
      v16h b = ldfrag(bp + g0, bp + g1);
      aco[nt] = WMMA(a, b, aco[nt]);
    }
  }
  #pragma unroll
  for (int nt = 0; nt < 8; ++nt) {
    float bv = b_out[nt * 16 + lr];
    #pragma unroll
    for (int r = 0; r < 8; ++r) {
      int row = rowBase + r + hs * 8;
      out[row * HID + nt * 16 + lr] = aco[nt][r] + bv;
    }
  }
}

// ---------------- launcher ----------------
extern "C" void kernel_launch(void* const* d_in, const int* in_sizes, int n_in,
                              void* d_out, int out_size, void* d_ws, size_t ws_size,
                              hipStream_t stream) {
  const float* x    = (const float*)d_in[0];
  const float* emb  = (const float*)d_in[1];
  const float* Wnum = (const float*)d_in[2];
  const float* Bnum = (const float*)d_in[3];
  const float* Win  = (const float*)d_in[4];
  const float* bin  = (const float*)d_in[5];
  const float* W1   = (const float*)d_in[6];
  const float* b1   = (const float*)d_in[7];
  const float* W2   = (const float*)d_in[8];
  const float* b2   = (const float*)d_in[9];
  const float* lng  = (const float*)d_in[10];
  const float* lnb  = (const float*)d_in[11];
  const float* Wout = (const float*)d_in[12];
  const float* bout = (const float*)d_in[13];
  _Float16* ws = (_Float16*)d_ws;
  float* out = (float*)d_out;

  int prepBlocks = (WS_TOTAL + 255) / 256;
  prep_kernel<<<prepBlocks, 256, 0, stream>>>(emb, Win, W1, W2, Wout, ws);

  int rows = in_sizes[0] / NFEAT;   // 98304
  int blocks = rows / 128;          // 768 workgroups of 8 waves
  embnn_main<<<blocks, 256, 0, stream>>>(x, Wnum, Bnum, bin, b1, b2, lng, lnb, bout, ws, out);
}
